// AttentionLayer_89541478187455
// MI455X (gfx1250) — matrix operand order repaired and checked by reading, not compiled, not run on hardware
//
#include <hip/hip_runtime.h>
#include <hip/hip_bf16.h>

typedef __attribute__((ext_vector_type(16))) __bf16 v16bf;
typedef __attribute__((ext_vector_type(8)))  __bf16 v8bf;
typedef __attribute__((ext_vector_type(4)))  __bf16 v4bf;
typedef __attribute__((ext_vector_type(8)))  float  v8f;

#define B_  2
#define S_  2048
#define D_  1024
#define H_  16
#define HD_ 64
#define M_  (B_*S_)

static __device__ __forceinline__ v16bf cat16(v8bf lo, v8bf hi) {
  return __builtin_shufflevector(lo, hi, 0,1,2,3,4,5,6,7,8,9,10,11,12,13,14,15);
}
static __device__ __forceinline__ v8f wmma_bf16(v16bf a, v16bf b, v8f c) {
  // 8 args: (neg_a, A, neg_b, B, c_mod, C, reuse_a, reuse_b)
  return __builtin_amdgcn_wmma_f32_16x16x32_bf16(false, a, false, b, (short)0, c, false, false);
}

// ---------------------------------------------------------------------------
// f32 -> bf16 conversion (vectorized x4)
// ---------------------------------------------------------------------------
__global__ __launch_bounds__(256)
void cvt_f32_bf16_v4(const float* __restrict__ src,
                     __bf16* __restrict__ dst, int n4) {
  int i = blockIdx.x * 256 + threadIdx.x;
  if (i < n4) {
    float4 f = ((const float4*)src)[i];
    v4bf o;
    o[0] = (__bf16)f.x; o[1] = (__bf16)f.y;
    o[2] = (__bf16)f.z; o[3] = (__bf16)f.w;
    ((v4bf*)dst)[i] = o;
  }
}

// ---------------------------------------------------------------------------
// C = A @ W^T + bias.  A:[M,K] bf16 row-major, W:[N,K] bf16 row-major.
// One wave per workgroup computes a 16x64 output tile (4 C fragments).
// MODE 0: bf16 row-major out   MODE 1: bf16 transposed ([b*D+n]*S + s) out
// MODE 2: f32 row-major out (final projection)
// ---------------------------------------------------------------------------
template<int MODE>
__global__ __launch_bounds__(32)
void gemm_xwT(const __bf16* __restrict__ A,
              const __bf16* __restrict__ W,
              const float*  __restrict__ bias,
              void* __restrict__ outp,
              int M, int N, int K) {
  const int lane = threadIdx.x & 31;
  const int hw   = lane >> 4;      // half-wave
  const int l16  = lane & 15;
  const int m0   = blockIdx.x * 16;
  const int n0   = blockIdx.y * 64;

  v8f acc[4] = {};

  const __bf16* arow  = A + (size_t)(m0 + l16) * K;
  const __bf16* wbase = W + (size_t)(n0 + l16) * K;

  for (int k0 = 0; k0 < K; k0 += 32) {
    // A fragment: lanes<16 hold K [k0..k0+15]; lanes>=16 hold [k0+16..k0+31]
    const __bf16* ap = arow + k0 + hw * 16;
    v16bf af = cat16(*(const v8bf*)ap, *(const v8bf*)(ap + 8));
#pragma unroll
    for (int t = 0; t < 4; ++t) {
      // B fragment: lane holds column n0+t*16+l16 = W row, contiguous 16 K-vals
      const __bf16* wp = wbase + (size_t)(t * 16) * K + k0 + hw * 16;
      v16bf wf = cat16(*(const v8bf*)wp, *(const v8bf*)(wp + 8));
      acc[t] = wmma_bf16(af, wf, acc[t]);
    }
  }

#pragma unroll
  for (int t = 0; t < 4; ++t) {
    int n = n0 + t * 16 + l16;
    float bv = bias[n];
#pragma unroll
    for (int r = 0; r < 8; ++r) {
      int m = m0 + r + hw * 8;                 // C layout: lanes>=16 -> rows +8
      float val = acc[t][r] + bv;
      if (MODE == 0) {
        ((__bf16*)outp)[(size_t)m * N + n] = (__bf16)val;
      } else if (MODE == 1) {
        size_t off = ((size_t)(m / S_) * N + n) * S_ + (size_t)(m % S_);
        ((__bf16*)outp)[off] = (__bf16)val;    // V stored transposed per (b,h)
      } else {
        ((float*)outp)[(size_t)m * N + n] = val;
      }
    }
  }
}

// ---------------------------------------------------------------------------
// Flash attention, fully transposed formulation:
//   S^T = K @ Q^T  (C rows = keys, C cols = queries -> one query per lane)
//   O^T = V^T @ P^T (C rows = head dims, C cols = queries)
// Per-lane softmax state (query == lane%16): no per-row arrays, one
// half-wave shuffle per reduction, scalar alpha rescale, packed stores.
// grid (B*H, S/64), block 128 (4 waves, each owns a 16-query tile).
// ---------------------------------------------------------------------------
__global__ __launch_bounds__(128)
void flash_attn(const __bf16* __restrict__ Qp,    // [B*S, D]
                const __bf16* __restrict__ Kp,    // [B*S, D]
                const __bf16* __restrict__ VT,    // [B*D, S]
                const int*    __restrict__ mask,  // [B, S]
                __bf16* __restrict__ ctx) {       // [B*S, D]
  const int bh   = blockIdx.x;
  const int b    = bh >> 4;
  const int h    = bh & 15;
  const int wave = threadIdx.x >> 5;
  const int lane = threadIdx.x & 31;
  const int hw   = lane >> 4;
  const int l16  = lane & 15;
  const int q0   = blockIdx.y * 64 + wave * 16;

  __shared__ __bf16 pbuf[4][16][40];   // per-wave P^T tile: [query][key], padded

  // Q^T as B operand: lane holds query row q0+l16, contiguous 16 d-values
  v16bf bq[2];
  const __bf16* qrow = Qp + (size_t)(b * S_ + q0 + l16) * D_ + h * HD_;
#pragma unroll
  for (int d = 0; d < 2; ++d) {
    const __bf16* p = qrow + d * 32 + hw * 16;
    bq[d] = cat16(*(const v8bf*)p, *(const v8bf*)(p + 8));
  }

  float mrun = -1e30f, lrun = 0.f;
  v8f   O[4] = {};                      // O^T: rows = head dims, cols = queries

  const float scale = 0.125f;           // 1/sqrt(64)
  const __bf16* kbase = Kp + (size_t)(b * S_) * D_ + h * HD_;
  const __bf16* vbase = VT + ((size_t)b * D_ + h * HD_) * S_;
  const int*    mrow  = mask + b * S_;

  for (int k0 = 0; k0 < S_; k0 += 32) {
    // prefetch next chunk (speculative; OOB prefetches are dropped)
    __builtin_prefetch(kbase + (size_t)(k0 + 32 + l16) * D_, 0, 1);
    __builtin_prefetch(vbase + (size_t)l16 * S_ + k0 + 32, 0, 1);

    // ---- S^T = K_chunk @ Q^T : 2 fragments (16 keys x 16 queries each) ----
    // element s[jt][r] belongs to key  k0 + jt*16 + hw*8 + r, query l16
    v8f s[2];
#pragma unroll
    for (int jt = 0; jt < 2; ++jt) {
      v8f sv = {};
#pragma unroll
      for (int d = 0; d < 2; ++d) {
        // A operand: lane holds key row k0+jt*16+l16, contiguous 16 d-values
        const __bf16* kp = kbase + (size_t)(k0 + jt * 16 + l16) * D_ + d * 32 + hw * 16;
        v16bf ak = cat16(*(const v8bf*)kp, *(const v8bf*)(kp + 8));
        sv = wmma_bf16(ak, bq[d], sv);
      }
      // mask per key (8 consecutive keys per half-wave -> two int4 loads)
      const int* mp = mrow + k0 + jt * 16 + hw * 8;
      int4 mv0 = *(const int4*)(mp);
      int4 mv1 = *(const int4*)(mp + 4);
      int mm[8] = { mv0.x, mv0.y, mv0.z, mv0.w, mv1.x, mv1.y, mv1.z, mv1.w };
#pragma unroll
      for (int r = 0; r < 8; ++r)
        sv[r] = mm[r] ? -1e30f : sv[r] * scale;
      s[jt] = sv;
    }

    // ---- online softmax: all 16 of this lane's scores are one query ----
    float m16 = -1e30f;
#pragma unroll
    for (int jt = 0; jt < 2; ++jt)
#pragma unroll
      for (int r = 0; r < 8; ++r) m16 = fmaxf(m16, s[jt][r]);
    m16 = fmaxf(m16, __shfl_xor(m16, 16, 32));   // combine the two key halves

    float mnew  = fmaxf(mrun, m16);
    float alpha = __expf(mrun - mnew);
    mrun = mnew;

    float rs = 0.f;
#pragma unroll
    for (int jt = 0; jt < 2; ++jt)
#pragma unroll
      for (int r = 0; r < 8; ++r) {
        float p = __expf(s[jt][r] - mnew);
        s[jt][r] = p;
        rs += p;
      }
    rs += __shfl_xor(rs, 16, 32);
    lrun = lrun * alpha + rs;

    // rescale running O^T: every element in this lane is query l16
#pragma unroll
    for (int t = 0; t < 4; ++t)
#pragma unroll
      for (int r = 0; r < 8; ++r) O[t][r] *= alpha;

    // ---- P^T -> LDS [query][key]; packed 16B stores ----
    __syncthreads();
#pragma unroll
    for (int jt = 0; jt < 2; ++jt) {
      v8bf pk;
#pragma unroll
      for (int r = 0; r < 8; ++r) pk[r] = (__bf16)s[jt][r];
      *(v8bf*)&pbuf[wave][l16][jt * 16 + hw * 8] = pk;
    }
    __syncthreads();
    // B operand for PV: lane = query l16, contiguous 16 keys at hw*16
    const __bf16* pr = &pbuf[wave][l16][hw * 16];
    v16bf pb = cat16(*(const v8bf*)pr, *(const v8bf*)(pr + 8));

    // ---- O^T += V^T_chunk @ P^T  (4 head-dim row tiles of 16) ----
#pragma unroll
    for (int t = 0; t < 4; ++t) {
      // A operand: lane holds V^T row (head dim t*16+l16), contiguous 16 keys
      const __bf16* vp = vbase + (size_t)(t * 16 + l16) * S_ + k0 + hw * 16;
      v16bf av = cat16(*(const v8bf*)vp, *(const v8bf*)(vp + 8));
      O[t] = wmma_bf16(av, pb, O[t]);
    }
  }

  // ---- normalize, store context: contiguous 16B stores per tile ----
  float inv_l = 1.0f / lrun;
  __bf16* crow = ctx + (size_t)(b * S_ + q0 + l16) * D_ + h * HD_;
#pragma unroll
  for (int t = 0; t < 4; ++t) {
    v8bf ov;
#pragma unroll
    for (int r = 0; r < 8; ++r) ov[r] = (__bf16)(O[t][r] * inv_l);
    // element r is head dim t*16 + hw*8 + r for query q0+l16
    *(v8bf*)(crow + t * 16 + hw * 8) = ov;
  }
}

// ---------------------------------------------------------------------------
extern "C" void kernel_launch(void* const* d_in, const int* in_sizes, int n_in,
                              void* d_out, int out_size, void* d_ws, size_t ws_size,
                              hipStream_t stream) {
  const float* q    = (const float*)d_in[0];
  const float* k    = (const float*)d_in[1];
  const float* v    = (const float*)d_in[2];
  const int*   mask = (const int*)  d_in[3];
  const float* Wq   = (const float*)d_in[4];
  const float* bq   = (const float*)d_in[5];
  const float* Wk   = (const float*)d_in[6];
  const float* bk   = (const float*)d_in[7];
  const float* Wv   = (const float*)d_in[8];
  const float* bv   = (const float*)d_in[9];
  const float* Wo   = (const float*)d_in[10];
  const float* bo   = (const float*)d_in[11];

  const size_t XN = (size_t)M_ * D_;   // 4M elems
  const size_t WN = (size_t)D_ * D_;   // 1M elems

  __bf16* ws  = (__bf16*)d_ws;
  __bf16* qb  = ws;
  __bf16* kb  = qb  + XN;
  __bf16* vb  = kb  + XN;
  __bf16* Wqb = vb  + XN;
  __bf16* Wkb = Wqb + WN;
  __bf16* Wvb = Wkb + WN;
  __bf16* Wob = Wvb + WN;
  __bf16* Qp  = Wob + WN;
  __bf16* Kp  = Qp  + XN;
  __bf16* VT  = Kp  + XN;
  __bf16* ctx = VT  + XN;     // total 32M bf16 = 64 MB

  // --- convert everything to bf16 once ---
  int xn4 = (int)(XN / 4), wn4 = (int)(WN / 4);
  cvt_f32_bf16_v4<<<(xn4 + 255) / 256, 256, 0, stream>>>(q,  qb,  xn4);
  cvt_f32_bf16_v4<<<(xn4 + 255) / 256, 256, 0, stream>>>(k,  kb,  xn4);
  cvt_f32_bf16_v4<<<(xn4 + 255) / 256, 256, 0, stream>>>(v,  vb,  xn4);
  cvt_f32_bf16_v4<<<(wn4 + 255) / 256, 256, 0, stream>>>(Wq, Wqb, wn4);
  cvt_f32_bf16_v4<<<(wn4 + 255) / 256, 256, 0, stream>>>(Wk, Wkb, wn4);
  cvt_f32_bf16_v4<<<(wn4 + 255) / 256, 256, 0, stream>>>(Wv, Wvb, wn4);
  cvt_f32_bf16_v4<<<(wn4 + 255) / 256, 256, 0, stream>>>(Wo, Wob, wn4);

  // --- projections ---
  dim3 gg(M_ / 16, D_ / 64);
  gemm_xwT<0><<<gg, 32, 0, stream>>>(qb, Wqb, bq, Qp, M_, D_, D_);
  gemm_xwT<0><<<gg, 32, 0, stream>>>(kb, Wkb, bk, Kp, M_, D_, D_);
  gemm_xwT<1><<<gg, 32, 0, stream>>>(vb, Wvb, bv, VT, M_, D_, D_);  // V transposed

  // --- flash attention ---
  flash_attn<<<dim3(B_ * H_, S_ / 64), 128, 0, stream>>>(Qp, Kp, VT, mask, ctx);

  // --- output projection (f32 out) ---
  gemm_xwT<2><<<gg, 32, 0, stream>>>(ctx, Wob, bo, d_out, M_, D_, D_);
}
